// MambaEncoder_36086315221484
// MI455X (gfx1250) — compile-verified, matrix operand order checked by reading, not run on hardware
//
#include <hip/hip_runtime.h>

typedef unsigned short bfraw;
typedef __attribute__((ext_vector_type(16))) __bf16 v16bf;
typedef __attribute__((ext_vector_type(8)))  float  v8f;
typedef __attribute__((ext_vector_type(8)))  unsigned short ushort8;

union FragAB { v16bf v; bfraw u[16]; ushort8 q[2]; };

static __device__ __forceinline__ bfraw f2bf(float f) {
    union { float f; unsigned int u; } x; x.f = f;
    unsigned int r = x.u + 0x7FFFu + ((x.u >> 16) & 1u);
    return (bfraw)(r >> 16);
}
static __device__ __forceinline__ float siluf(float v) { return v / (1.f + __expf(-v)); }
static __device__ __forceinline__ float lreluf(float v) { return v >= 0.f ? v : 0.01f * v; }

// ---------------------------------------------------------------------------
// WMMA GEMM: C(MxN) f32 = A(MxK) bf16 row-major x Bt(NxK) bf16 row-major (B^T)
// One wave32 computes a 16x32 tile (A fragment reused over two B tiles).
// Per-lane fragment data is contiguous along K -> b128 vector loads.
// Out-of-range rows/cols are CLAMPED on load (affects only OOB outputs),
// so the hot loop has no exec-mask manipulation; stores are guarded.
// biasMode: 0 none, 1 per-row (bias[M]), 2 per-col (bias[N])
// ---------------------------------------------------------------------------
__global__ __launch_bounds__(256) void k_wmma_gemm(
    const bfraw* __restrict__ A, const bfraw* __restrict__ Bt,
    const float* __restrict__ bias, int biasMode, float* __restrict__ C,
    int M, int N, int K)
{
    int waveId = (blockIdx.x * blockDim.x + threadIdx.x) >> 5;
    int lane   = threadIdx.x & 31;
    int tilesM = (M + 15) >> 4;
    int tilesN = (N + 31) >> 5;            // 32 columns per wave
    if (waveId >= tilesM * tilesN) return;
    int tileM = (waveId / tilesN) << 4;
    int tileN = (waveId % tilesN) << 5;

    int half = lane >> 4;                  // K-half selector per 16-bit A/B layout
    int lmod = lane & 15;
    int arow  = tileM + lmod;        if (arow  > M - 1) arow  = M - 1;
    int bcol0 = tileN + lmod;        if (bcol0 > N - 1) bcol0 = N - 1;
    int bcol1 = tileN + 16 + lmod;   if (bcol1 > N - 1) bcol1 = N - 1;

    const bfraw* Ar  = A  + (size_t)arow  * K;
    const bfraw* Br0 = Bt + (size_t)bcol0 * K;
    const bfraw* Br1 = Bt + (size_t)bcol1 * K;

    v8f acc0 = {}, acc1 = {};
    int kvec = ((K & 7) == 0) ? (K & ~31) : 0;   // vector path needs 16B alignment

    for (int k0 = 0; k0 < kvec; k0 += 32) {
        int kb = k0 + half * 8;
        FragAB fa, fb0, fb1;
        fa.q[0]  = *(const ushort8*)(Ar  + kb);
        fa.q[1]  = *(const ushort8*)(Ar  + kb + 16);
        fb0.q[0] = *(const ushort8*)(Br0 + kb);
        fb0.q[1] = *(const ushort8*)(Br0 + kb + 16);
        fb1.q[0] = *(const ushort8*)(Br1 + kb);
        fb1.q[1] = *(const ushort8*)(Br1 + kb + 16);
        if (k0 + 32 < kvec) {               // speculative: global_prefetch_b8
            __builtin_prefetch(Ar  + kb + 32, 0, 1);
            __builtin_prefetch(Br0 + kb + 32, 0, 1);
            __builtin_prefetch(Br1 + kb + 32, 0, 1);
        }
        acc0 = __builtin_amdgcn_wmma_f32_16x16x32_bf16(
            false, fa.v, false, fb0.v, (short)0, acc0, false, false);
        acc1 = __builtin_amdgcn_wmma_f32_16x16x32_bf16(
            false, fa.v, false, fb1.v, (short)0, acc1, false, false);
    }
    // masked tail (clamped loads + select; no exec branching)
    for (int k0 = kvec; k0 < K; k0 += 32) {
        FragAB fa, fb0, fb1;
#pragma unroll
        for (int j = 0; j < 8; ++j) {
            int k1 = k0 + half * 8 + j;
            int k2 = k1 + 16;
            int k1c = k1 < K - 1 ? k1 : K - 1;
            int k2c = k2 < K - 1 ? k2 : K - 1;
            bfraw a1 = Ar[k1c],  a2 = Ar[k2c];
            bfraw b1 = Br0[k1c], b2 = Br0[k2c];
            bfraw c1 = Br1[k1c], c2 = Br1[k2c];
            fa.u[j]      = (k1 < K) ? a1 : (bfraw)0;
            fa.u[j + 8]  = (k2 < K) ? a2 : (bfraw)0;
            fb0.u[j]     = (k1 < K) ? b1 : (bfraw)0;
            fb0.u[j + 8] = (k2 < K) ? b2 : (bfraw)0;
            fb1.u[j]     = (k1 < K) ? c1 : (bfraw)0;
            fb1.u[j + 8] = (k2 < K) ? c2 : (bfraw)0;
        }
        acc0 = __builtin_amdgcn_wmma_f32_16x16x32_bf16(
            false, fa.v, false, fb0.v, (short)0, acc0, false, false);
        acc1 = __builtin_amdgcn_wmma_f32_16x16x32_bf16(
            false, fa.v, false, fb1.v, (short)0, acc1, false, false);
    }
    // store (guarded)
    int colx0 = tileN + lmod, colx1 = tileN + 16 + lmod;
#pragma unroll
    for (int r = 0; r < 8; ++r) {
        int row = tileM + r + 8 * half;    // C/D layout: VGPR r, lanes>=16 => M+8
        if (row >= M) continue;
        float v0 = acc0[r], v1 = acc1[r];
        if (biasMode == 1) { v0 += bias[row]; v1 += bias[row]; }
        if (colx0 < N) C[(size_t)row * N + colx0] = v0 + (biasMode == 2 ? bias[colx0] : 0.f);
        if (colx1 < N) C[(size_t)row * N + colx1] = v1 + (biasMode == 2 ? bias[colx1] : 0.f);
    }
}

// ---------------------------------------------------------------------------
// Direct conv3d (downsample convs)
// ---------------------------------------------------------------------------
__global__ void k_conv3d(const float* __restrict__ x, const float* __restrict__ w,
                         const float* __restrict__ bias, float* __restrict__ out,
                         int Cin, int Di, int Hi, int Wi,
                         int Cout, int Do, int Ho, int Wo,
                         int ks, int stride, int pad)
{
    int idx = blockIdx.x * blockDim.x + threadIdx.x;
    int total = Cout * Do * Ho * Wo;
    if (idx >= total) return;
    int wo = idx % Wo, t = idx / Wo;
    int ho = t % Ho;  t /= Ho;
    int dz = t % Do;  int co = t / Do;
    float s = bias[co];
    for (int ci = 0; ci < Cin; ++ci)
        for (int kd = 0; kd < ks; ++kd) {
            int zi = dz * stride - pad + kd;
            if (zi < 0 || zi >= Di) continue;
            for (int kh = 0; kh < ks; ++kh) {
                int yi = ho * stride - pad + kh;
                if (yi < 0 || yi >= Hi) continue;
                for (int kw = 0; kw < ks; ++kw) {
                    int xi = wo * stride - pad + kw;
                    if (xi < 0 || xi >= Wi) continue;
                    s += w[(((co * Cin + ci) * ks + kd) * ks + kh) * ks + kw]
                       * x[((ci * Di + zi) * Hi + yi) * Wi + xi];
                }
            }
        }
    out[idx] = s;
}

// im2col 3x3x3 pad1 stride1: x:(C,D,H,W) -> col:(L, C*27) bf16  (Bt layout)
__global__ void k_im2col3(const float* __restrict__ x, bfraw* __restrict__ col,
                          int C, int D, int H, int W)
{
    int L = D * H * W;
    int K27 = C * 27;
    int idx = blockIdx.x * blockDim.x + threadIdx.x;
    if (idx >= L * K27) return;
    int k = idx % K27, l = idx / K27;
    int c = k / 27, j = k % 27;
    int kd = j / 9, kh = (j / 3) % 3, kw = j % 3;
    int wz = l % W, hz = (l / W) % H, dz = l / (W * H);
    int zd = dz - 1 + kd, zh = hz - 1 + kh, zw = wz - 1 + kw;
    float v = 0.f;
    if (zd >= 0 && zd < D && zh >= 0 && zh < H && zw >= 0 && zw < W)
        v = x[((c * D + zd) * H + zh) * W + zw];
    col[idx] = f2bf(v);
}

// instance norm over L per channel; optional fused lrelu; out may alias in
__global__ __launch_bounds__(256) void k_instnorm(
    const float* __restrict__ in, float* __restrict__ out, int C, int L, int dolrelu)
{
    __shared__ float s1[256], s2[256];
    int c = blockIdx.x;
    float a = 0.f, b = 0.f;
    for (int l = threadIdx.x; l < L; l += blockDim.x) {
        float v = in[(size_t)c * L + l]; a += v; b += v * v;
    }
    s1[threadIdx.x] = a; s2[threadIdx.x] = b; __syncthreads();
    for (int o = 128; o > 0; o >>= 1) {
        if (threadIdx.x < o) { s1[threadIdx.x] += s1[threadIdx.x + o]; s2[threadIdx.x] += s2[threadIdx.x + o]; }
        __syncthreads();
    }
    float mu = s1[0] / L;
    float var = s2[0] / L - mu * mu;
    float rs = rsqrtf(var + 1e-5f);
    for (int l = threadIdx.x; l < L; l += blockDim.x) {
        float v = (in[(size_t)c * L + l] - mu) * rs;
        if (dolrelu) v = lreluf(v);
        out[(size_t)c * L + l] = v;
    }
}

// layer norm over channels reading (C,L), writing bf16 (L,C)
__global__ void k_layernorm_t(const float* __restrict__ x, const float* __restrict__ g,
                              const float* __restrict__ b, bfraw* __restrict__ out,
                              int C, int L)
{
    int l = blockIdx.x * blockDim.x + threadIdx.x;
    if (l >= L) return;
    float mu = 0.f;
    for (int c = 0; c < C; ++c) mu += x[(size_t)c * L + l];
    mu /= C;
    float var = 0.f;
    for (int c = 0; c < C; ++c) { float d = x[(size_t)c * L + l] - mu; var += d * d; }
    var /= C;
    float rs = rsqrtf(var + 1e-5f);
    for (int c = 0; c < C; ++c)
        out[(size_t)l * C + c] = f2bf((x[(size_t)c * L + l] - mu) * rs * g[c] + b[c]);
}

// out(L,Wd) = in(src(l),:)  mode: 1=reverse, 2=slice perm(ns)
__global__ void k_permute_rows(const float* __restrict__ in, float* __restrict__ out,
                               int L, int Wd, int mode, int ns)
{
    int idx = blockIdx.x * blockDim.x + threadIdx.x;
    if (idx >= L * Wd) return;
    int l = idx / Wd, c = idx % Wd;
    int ls = (mode == 1) ? (L - 1 - l) : ((l % ns) * (L / ns) + l / ns);
    out[(size_t)l * Wd + c] = in[(size_t)ls * Wd + c];
}

// causal depthwise conv1d (k=4, left pad 3) + SiLU; xz:(L,2*di), ch [0,di)
__global__ void k_conv1d_silu(const float* __restrict__ xz, const float* __restrict__ w,
                              const float* __restrict__ b, float* __restrict__ xt,
                              bfraw* __restrict__ xtb, int L, int di)
{
    int idx = blockIdx.x * blockDim.x + threadIdx.x;
    if (idx >= L * di) return;
    int l = idx / di, d = idx % di;
    float s = b[d];
#pragma unroll
    for (int j = 0; j < 4; ++j) {
        int ls = l - 3 + j;
        if (ls >= 0) s += w[d * 4 + j] * xz[(size_t)ls * (2 * di) + d];
    }
    float y = siluf(s);
    xt[idx] = y;
    xtb[idx] = f2bf(y);
}

__global__ void k_softplus_bias(const float* __restrict__ in, const float* __restrict__ bias,
                                float* __restrict__ out, int n, int di)
{
    int idx = blockIdx.x * blockDim.x + threadIdx.x;
    if (idx >= n) return;
    float v = in[idx] + bias[idx % di];
    out[idx] = (v > 20.f) ? v : log1pf(__expf(v));
}

// fused selective scan + C-einsum + D skip + z gating; writes through inverse perm
__global__ void k_ssm_scan(const float* __restrict__ xt, const float* __restrict__ delta,
                           const float* __restrict__ xdbl, int R,
                           const float* __restrict__ Alog, const float* __restrict__ Dp,
                           const float* __restrict__ xz, float* __restrict__ ysum,
                           int L, int di, int accumulate, int mode, int ns)
{
    int d = blockIdx.x * blockDim.x + threadIdx.x;
    if (d >= di) return;
    float A[16], h[16];
#pragma unroll
    for (int n = 0; n < 16; ++n) { A[n] = -__expf(Alog[d * 16 + n]); h[n] = 0.f; }
    float Dv = Dp[d];
    int Wd = R + 32;
    for (int l = 0; l < L; ++l) {
        float del = delta[(size_t)l * di + d];
        float xv  = xt[(size_t)l * di + d];
        float acc = 0.f;
#pragma unroll
        for (int n = 0; n < 16; ++n) {
            float Bv = xdbl[(size_t)l * Wd + R + n];
            float Cv = xdbl[(size_t)l * Wd + R + 16 + n];
            float dA = __expf(del * A[n]);
            h[n] = dA * h[n] + del * Bv * xv;
            acc += h[n] * Cv;
        }
        float zv = xz[(size_t)l * (2 * di) + di + d];
        float y = (acc + xv * Dv) * siluf(zv);
        int lo = l;
        if (mode == 1) lo = L - 1 - l;
        else if (mode == 2) lo = (l % ns) * (L / ns) + l / ns;
        if (accumulate) ysum[(size_t)lo * di + d] += y;
        else            ysum[(size_t)lo * di + d]  = y;
    }
}

__global__ void k_cvt_bf16(const float* __restrict__ in, bfraw* __restrict__ out, int n)
{
    int idx = blockIdx.x * blockDim.x + threadIdx.x;
    if (idx < n) out[idx] = f2bf(in[idx]);
}

// out (L,R) bf16 = first R cols of in (L,Wd)
__global__ void k_slice_bf16(const float* __restrict__ in, bfraw* __restrict__ out,
                             int L, int Wd, int R)
{
    int idx = blockIdx.x * blockDim.x + threadIdx.x;
    if (idx >= L * R) return;
    int l = idx / R, r = idx % R;
    out[idx] = f2bf(in[(size_t)l * Wd + r]);
}

__global__ void k_add_lrelu(const float* __restrict__ a, const float* __restrict__ b,
                            float* __restrict__ out, int n)
{
    int idx = blockIdx.x * blockDim.x + threadIdx.x;
    if (idx < n) out[idx] = lreluf(a[idx] + b[idx]);
}

// x(C,L) += y(L,C)
__global__ void k_add_transposed(float* __restrict__ x, const float* __restrict__ y,
                                 int C, int L)
{
    int idx = blockIdx.x * blockDim.x + threadIdx.x;
    if (idx >= C * L) return;
    int c = idx / L, l = idx % L;
    x[idx] += y[(size_t)l * C + c];
}

static inline size_t mx(size_t a, size_t b) { return a > b ? a : b; }

extern "C" void kernel_launch(void* const* d_in, const int* in_sizes, int n_in,
                              void* d_out, int out_size, void* d_ws, size_t ws_size,
                              hipStream_t stream)
{
    (void)in_sizes; (void)n_in; (void)out_size; (void)ws_size;
    const float* x_in = (const float*)d_in[0];
    int cur = 1;
    auto P = [&]() -> const float* { return (const float*)d_in[cur++]; };

    static const int DM[4] = {48, 96, 192, 384};
    static const int NS[4] = {48, 24, 12, 6};
    static const int Dsp[4] = {24, 12, 6, 3};
    static const int Hsp[4] = {32, 16, 8, 4};
    static const int Wsp[4] = {32, 16, 8, 4};

    char* base = (char*)d_ws;
    size_t off = 0;
    auto alloc = [&](size_t bytes) -> void* {
        off = (off + 255) & ~(size_t)255;
        void* p = base + off;
        off += bytes;
        return p;
    };

    size_t mCL = 0, mKL = 0, mWK = 0, mXZ = 0, mLD = 0, mLDI = 0, mLW = 0, mLR = 0;
    size_t mBip = 0, mBxp = 0, mBdt = 0, mBop = 0;
    for (int i = 0; i < 4; ++i) {
        size_t L = (size_t)Dsp[i] * Hsp[i] * Wsp[i], C = DM[i], di = 2 * C, R = (C + 15) / 16;
        mCL = mx(mCL, C * L);             mKL = mx(mKL, C * 27 * L);
        mWK = mx(mWK, C * C * 27);        mXZ = mx(mXZ, L * 2 * di);
        mLD = mx(mLD, L * C);             mLDI = mx(mLDI, L * di);
        mLW = mx(mLW, L * (R + 32));      mLR = mx(mLR, L * R);
        mBip = mx(mBip, C * 4 * C);       mBxp = mx(mBxp, di * (R + 32));
        mBdt = mx(mBdt, R * di);          mBop = mx(mBop, di * C);
    }
    float* xbuf  = (float*)alloc(mCL * 4);
    float* ybuf  = (float*)alloc(mCL * 4);
    float* y2buf = (float*)alloc(mCL * 4);
    bfraw* col   = (bfraw*)alloc(mKL * 2);
    bfraw* wbf   = (bfraw*)alloc(mWK * 2);
    float* xz    = (float*)alloc(mXZ * 4);
    float* xzp   = (float*)alloc(mXZ * 4);
    bfraw* xnb   = (bfraw*)alloc(mLD * 2);
    bfraw* Bip   = (bfraw*)alloc(mBip * 2);
    float* xt    = (float*)alloc(mLDI * 4);
    bfraw* xtb   = (bfraw*)alloc(mLDI * 2);
    bfraw* Bxp   = (bfraw*)alloc(mBxp * 2);
    float* xdbl  = (float*)alloc(mLW * 4);
    bfraw* dtb   = (bfraw*)alloc(mLR * 2);
    bfraw* Bdt   = (bfraw*)alloc(mBdt * 2);
    float* dtg   = (float*)alloc(mLDI * 4);
    float* delta = (float*)alloc(mLDI * 4);
    float* ysum  = (float*)alloc(mLDI * 4);
    bfraw* ysb   = (bfraw*)alloc(mLDI * 2);
    bfraw* Bop   = (bfraw*)alloc(mBop * 2);
    float* xm    = (float*)alloc(mLD * 4);

    auto gemm = [&](const bfraw* A, const bfraw* Bt, const float* bias, int biasMode,
                    float* C, int M, int N, int K) {
        int tiles = ((M + 15) / 16) * ((N + 31) / 32);
        int blocks = (tiles * 32 + 255) / 256;
        k_wmma_gemm<<<blocks, 256, 0, stream>>>(A, Bt, bias, biasMode, C, M, N, K);
    };
#define EW(n) <<<((n) + 255) / 256, 256, 0, stream>>>

    float* outp = (float*)d_out;
    size_t outoff = 0;
    const float* prev_out = nullptr;
    int pC = 1, pD = 48, pH = 64, pW = 64;

    for (int st = 0; st < 4; ++st) {
        const int C = DM[st], D = Dsp[st], H = Hsp[st], W = Wsp[st];
        const int L = D * H * W, di = 2 * C, R = (C + 15) / 16, Wd = R + 32, ns = NS[st];

        // ---- params (flattened dict order) ----
        const float* down_w = P(); const float* down_b = P();
        const float* w1 = P(); const float* b1 = P();
        const float* w2 = P(); const float* b2 = P();
        const float* ln_g = P(); const float* ln_b = P();
        const float* ipw = P(); const float* opw = P();
        struct Br { const float *cw, *cb, *xpw, *dpw, *dpb, *Al, *Dp; } br[3];
        for (int j = 0; j < 3; ++j) {
            br[j].cw = P(); br[j].cb = P(); br[j].xpw = P();
            br[j].dpw = P(); br[j].dpb = P(); br[j].Al = P(); br[j].Dp = P();
        }

        // ---- down conv ----
        const float* sin = (st == 0) ? x_in : prev_out;
        int ks = (st == 0) ? 7 : 2, pad = (st == 0) ? 3 : 0;
        k_conv3d EW(C * L)(sin, down_w, down_b, xbuf, pC, pD, pH, pW, C, D, H, W, ks, 2, pad);

        // ---- res block via im2col + WMMA GEMM ----
        int K27 = C * 27;
        k_im2col3 EW(L * K27)(xbuf, col, C, D, H, W);
        k_cvt_bf16 EW(C * K27)(w1, wbf, C * K27);
        gemm(wbf, col, b1, 1, ybuf, C, L, K27);       // A=(C,K27), Bt=col=(L,K27)
        k_instnorm<<<C, 256, 0, stream>>>(ybuf, ybuf, C, L, 1);
        k_im2col3 EW(L * K27)(ybuf, col, C, D, H, W);
        k_cvt_bf16 EW(C * K27)(w2, wbf, C * K27);
        gemm(wbf, col, b2, 1, y2buf, C, L, K27);
        k_instnorm<<<C, 256, 0, stream>>>(y2buf, y2buf, C, L, 0);
        k_add_lrelu EW(C * L)(y2buf, xbuf, xbuf, C * L);

        // ---- mamba layer ----
        k_layernorm_t EW(L)(xbuf, ln_g, ln_b, xnb, C, L);
        k_cvt_bf16 EW(4 * C * C)(ipw, Bip, 4 * C * C);     // in_proj_w (4dm,dm) = Bt directly
        gemm(xnb, Bip, nullptr, 0, xz, L, 4 * C, C);       // xz (L, 2*di)

        for (int b2i = 0; b2i < 3; ++b2i) {
            int mode = b2i;                                 // 0=f, 1=b, 2=s
            const float* xzs = xz;
            if (mode != 0) {
                k_permute_rows EW(L * 2 * di)(xz, xzp, L, 2 * di, mode, ns);
                xzs = xzp;
            }
            k_conv1d_silu EW(L * di)(xzs, br[b2i].cw, br[b2i].cb, xt, xtb, L, di);
            k_cvt_bf16 EW(Wd * di)(br[b2i].xpw, Bxp, Wd * di);   // x_proj_w (Wd,di) = Bt
            gemm(xtb, Bxp, nullptr, 0, xdbl, L, Wd, di);
            k_slice_bf16 EW(L * R)(xdbl, dtb, L, Wd, R);
            k_cvt_bf16 EW(di * R)(br[b2i].dpw, Bdt, di * R);     // dt_proj_w (di,R) = Bt
            gemm(dtb, Bdt, nullptr, 0, dtg, L, di, R);
            k_softplus_bias EW(L * di)(dtg, br[b2i].dpb, delta, L * di, di);
            k_ssm_scan<<<(di + 127) / 128, 128, 0, stream>>>(
                xt, delta, xdbl, R, br[b2i].Al, br[b2i].Dp, xzs, ysum,
                L, di, (b2i != 0) ? 1 : 0, mode, ns);
        }
        k_cvt_bf16 EW(L * di)(ysum, ysb, L * di);
        k_cvt_bf16 EW(C * di)(opw, Bop, C * di);                 // out_proj_w (dm,di) = Bt
        gemm(ysb, Bop, nullptr, 0, xm, L, C, di);
        k_add_transposed EW(C * L)(xbuf, xm, C, L);              // residual

        // ---- stage output: instance norm into d_out (also next stage input) ----
        k_instnorm<<<C, 256, 0, stream>>>(xbuf, outp + outoff, C, L, 0);
        prev_out = outp + outoff;
        outoff += (size_t)C * L;
        pC = C; pD = D; pH = H; pW = W;
    }
#undef EW
}
